// RnaMoeLayer_58402965291220
// MI455X (gfx1250) — matrix operand order repaired and checked by reading, not compiled, not run on hardware
//
#include <hip/hip_runtime.h>
#include <math.h>

typedef __bf16 bf16;
typedef __attribute__((ext_vector_type(16))) __bf16 v16bf;
typedef __attribute__((ext_vector_type(8)))  __bf16 v8bf;
typedef __attribute__((ext_vector_type(8)))  float  v8f;
typedef __attribute__((ext_vector_type(4)))  int    v4i;

constexpr int NB = 2, NS = 1024, ND = 768, NH = 12, NDH = 64, NE = 8, NI = 2048;
constexpr int NT = NB * NS;           // 2048 tokens
constexpr int MAXP = 1026;            // MAXPOS

// macro-tile geometry for the big GEMMs
constexpr int BM = 128, BN = 64, BK = 32;
constexpr int AP = 40;                // LDS row pitch (elems): 80B = 16B-aligned, 20-bank stride

#if defined(__has_builtin)
#if __has_builtin(__builtin_amdgcn_global_load_async_to_lds_b128)
#define ASYNC_CP 1
#endif
#if __has_builtin(__builtin_amdgcn_s_wait_asynccnt)
#define HAVE_WAIT_ASYNC 1
#endif
#endif
#ifndef ASYNC_CP
#define ASYNC_CP 0
#endif
#ifndef HAVE_WAIT_ASYNC
#define HAVE_WAIT_ASYNC 0
#endif

__device__ __forceinline__ void cp16(bf16* lds, const bf16* gl) {
#if ASYNC_CP
  typedef __attribute__((address_space(1))) v4i* gp_t;
  typedef __attribute__((address_space(3))) v4i* lp_t;
  __builtin_amdgcn_global_load_async_to_lds_b128((gp_t)gl, (lp_t)lds, 0, 0);
#else
  *(v8bf*)lds = *(const v8bf*)gl;
#endif
}

template <int N>
__device__ __forceinline__ void wait_async() {
#if HAVE_WAIT_ASYNC
  __builtin_amdgcn_s_wait_asynccnt(N);
#else
  asm volatile("s_wait_asynccnt %0" ::"i"(N) : "memory");
#endif
}

__device__ __forceinline__ int wave_id() {
  return blockIdx.x * (blockDim.x >> 5) + (threadIdx.x >> 5);
}

__device__ __forceinline__ v8f wmma_bf16(v16bf a, v16bf b, v8f c) {
  return __builtin_amdgcn_wmma_f32_16x16x32_bf16(false, a, false, b, (short)0, c, false, false);
}

// ---- global-direct fragment loaders (ISA 7.12.2 layouts, bf16 16x16x32) ----
__device__ __forceinline__ v16bf load_a_frag(const bf16* __restrict__ A, int lda,
                                             int m0, int k0, int lane) {
  const int m  = m0 + (lane & 15);
  const int kb = k0 + ((lane & 16) ? 8 : 0);
  const bf16* p = A + (size_t)m * lda + kb;
  __builtin_prefetch((const void*)(p + 32), 0, 0);
  v16bf a;
#pragma unroll
  for (int j = 0; j < 8; ++j) { a[j] = p[j]; a[j + 8] = p[j + 16]; }
  return a;
}
__device__ __forceinline__ v16bf load_b_frag(const bf16* __restrict__ W, int ldw,
                                             int n0, int k0, int lane) {
  const int n  = n0 + (lane & 15);
  const int kb = k0 + ((lane & 16) ? 8 : 0);
  const bf16* p = W + (size_t)n * ldw + kb;
  v16bf b;
#pragma unroll
  for (int j = 0; j < 8; ++j) { b[j] = p[j]; b[j + 8] = p[j + 16]; }
  return b;
}

// ---- LDS fragment loaders (pitch AP) ----
__device__ __forceinline__ v16bf lds_a_frag(const bf16* S, int m0, int lane) {
  const bf16* p = S + (m0 + (lane & 15)) * AP + ((lane & 16) ? 8 : 0);
  v16bf a;
#pragma unroll
  for (int j = 0; j < 8; ++j) { a[j] = p[j]; a[j + 8] = p[j + 16]; }
  return a;
}
__device__ __forceinline__ v16bf lds_b_frag(const bf16* S, int n0, int lane) {
  const bf16* p = S + (n0 + (lane & 15)) * AP + ((lane & 16) ? 8 : 0);
  v16bf b;
#pragma unroll
  for (int j = 0; j < 8; ++j) { b[j] = p[j]; b[j + 8] = p[j + 16]; }
  return b;
}

// ================= tiled GEMM: C[M,N] = A@W^T + bias =================
// block: 256 thr (8 waves); macro tile 128x64; wave tile 32x32 (2x2 WMMA);
// double-buffered async global->LDS staging.
__global__ __launch_bounds__(256) void k_gemm_tiled(
    const bf16* __restrict__ A, int lda, const bf16* __restrict__ W, int ldw,
    float* __restrict__ C, int ldc, const float* __restrict__ bias,
    int nMacro, int K) {
  __shared__ bf16 sA[2][BM * AP];
  __shared__ bf16 sB[2][BN * AP];
  const int tid = threadIdx.x;
  const int m0 = (blockIdx.x / nMacro) * BM;
  const int n0 = (blockIdx.x % nMacro) * BN;

  auto stage = [&](int kt, int buf) {
    const int k0 = kt * BK;
#pragma unroll
    for (int i = 0; i < 2; ++i) {                 // A: 512 x 16B chunks
      const int c = tid + (i << 8);
      const int r = c >> 2, cc = (c & 3) << 3;
      cp16(&sA[buf][r * AP + cc], A + (size_t)(m0 + r) * lda + k0 + cc);
    }
    {                                             // B: 256 x 16B chunks
      const int r = tid >> 2, cc = (tid & 3) << 3;
      cp16(&sB[buf][r * AP + cc], W + (size_t)(n0 + r) * ldw + k0 + cc);
    }
  };

  const int lane = tid & 31, w = tid >> 5;
  const int wm = (w & 3) << 5, wn = (w >> 2) << 5;
  v8f acc[2][2] = {};
  const int KT = K / BK;

  stage(0, 0);
  for (int kt = 0; kt < KT; ++kt) {
    const int cur = kt & 1;
    if (kt + 1 < KT) { stage(kt + 1, cur ^ 1); wait_async<3>(); }
    else             { wait_async<0>(); }
    __syncthreads();
    v16bf b0 = lds_b_frag(sB[cur], wn,      lane);
    v16bf b1 = lds_b_frag(sB[cur], wn + 16, lane);
#pragma unroll
    for (int i = 0; i < 2; ++i) {
      v16bf a = lds_a_frag(sA[cur], wm + (i << 4), lane);
      acc[i][0] = wmma_bf16(a, b0, acc[i][0]);
      acc[i][1] = wmma_bf16(a, b1, acc[i][1]);
    }
    __syncthreads();
  }
#pragma unroll
  for (int i = 0; i < 2; ++i) {
    const int mo = m0 + wm + (i << 4) + ((lane & 16) ? 8 : 0);
#pragma unroll
    for (int j = 0; j < 2; ++j) {
      const int n = n0 + wn + (j << 4) + (lane & 15);
      const float bb = bias ? bias[n] : 0.0f;
#pragma unroll
      for (int g = 0; g < 8; ++g) C[(size_t)(mo + g) * ldc + n] = acc[i][j][g] + bb;
    }
  }
}

// ===== fused expert up-proj (tiled): g = gelu(ao@W1^T) * (ao@W2^T) =====
__global__ __launch_bounds__(256) void k_moe_h_tiled(
    const bf16* __restrict__ A, const bf16* __restrict__ W1,
    const bf16* __restrict__ W2, bf16* __restrict__ gout) {
  __shared__ bf16 sA[2][BM * AP];
  __shared__ bf16 sB1[2][BN * AP];
  __shared__ bf16 sB2[2][BN * AP];
  const int tid = threadIdx.x;
  constexpr int nMacro = NI / BN;                 // 32
  const int m0 = (blockIdx.x / nMacro) * BM;
  const int n0 = (blockIdx.x % nMacro) * BN;

  auto stage = [&](int kt, int buf) {
    const int k0 = kt * BK;
#pragma unroll
    for (int i = 0; i < 2; ++i) {
      const int c = tid + (i << 8);
      const int r = c >> 2, cc = (c & 3) << 3;
      cp16(&sA[buf][r * AP + cc], A + (size_t)(m0 + r) * ND + k0 + cc);
    }
    const int r = tid >> 2, cc = (tid & 3) << 3;
    cp16(&sB1[buf][r * AP + cc], W1 + (size_t)(n0 + r) * ND + k0 + cc);
    cp16(&sB2[buf][r * AP + cc], W2 + (size_t)(n0 + r) * ND + k0 + cc);
  };

  const int lane = tid & 31, w = tid >> 5;
  const int wm = (w & 3) << 5, wn = (w >> 2) << 5;
  v8f acc1[2][2] = {}, acc2[2][2] = {};
  constexpr int KT = ND / BK;                     // 24

  stage(0, 0);
  for (int kt = 0; kt < KT; ++kt) {
    const int cur = kt & 1;
    if (kt + 1 < KT) { stage(kt + 1, cur ^ 1); wait_async<4>(); }
    else             { wait_async<0>(); }
    __syncthreads();
    v16bf p0 = lds_b_frag(sB1[cur], wn,      lane);
    v16bf p1 = lds_b_frag(sB1[cur], wn + 16, lane);
    v16bf q0 = lds_b_frag(sB2[cur], wn,      lane);
    v16bf q1 = lds_b_frag(sB2[cur], wn + 16, lane);
#pragma unroll
    for (int i = 0; i < 2; ++i) {
      v16bf a = lds_a_frag(sA[cur], wm + (i << 4), lane);
      acc1[i][0] = wmma_bf16(a, p0, acc1[i][0]);
      acc1[i][1] = wmma_bf16(a, p1, acc1[i][1]);
      acc2[i][0] = wmma_bf16(a, q0, acc2[i][0]);
      acc2[i][1] = wmma_bf16(a, q1, acc2[i][1]);
    }
    __syncthreads();
  }
#pragma unroll
  for (int i = 0; i < 2; ++i) {
    const int mo = m0 + wm + (i << 4) + ((lane & 16) ? 8 : 0);
#pragma unroll
    for (int j = 0; j < 2; ++j) {
      const int n = n0 + wn + (j << 4) + (lane & 15);
#pragma unroll
      for (int g = 0; g < 8; ++g) {
        const float x = acc1[i][j][g];
        const float ge = 0.5f * x * (1.0f + erff(x * 0.70710678f));
        gout[(size_t)(mo + g) * NI + n] = (bf16)(ge * acc2[i][j][g]);
      }
    }
  }
}

// ---------- scores[b,h,q,k] = q . k (per-head, K=64) ----------
__global__ void k_scores(const bf16* __restrict__ qb, const bf16* __restrict__ kb,
                         float* __restrict__ sc, int tilesTotal) {
  int t = wave_id();
  if (t >= tilesTotal) return;
  const int nt = t & 63; t >>= 6;
  const int mt = t & 63; const int bh = t >> 6;
  const int b = bh / NH, h = bh % NH;
  const bf16* A = qb + (size_t)b * NS * ND + h * NDH;
  const bf16* W = kb + (size_t)b * NS * ND + h * NDH;
  float* C = sc + (size_t)bh * NS * NS;
  const int lane = threadIdx.x & 31;
  v8f acc = {};
  for (int k0 = 0; k0 < NDH; k0 += 32) {
    v16bf a  = load_a_frag(A, ND, mt << 4, k0, lane);
    v16bf bb = load_b_frag(W, ND, nt << 4, k0, lane);
    acc = wmma_bf16(a, bb, acc);
  }
  const int n  = (nt << 4) + (lane & 15);
  const int mo = (mt << 4) + ((lane & 16) ? 8 : 0);
#pragma unroll
  for (int g = 0; g < 8; ++g) C[(size_t)(mo + g) * NS + n] = acc[g];
}

// ---------- relative-position term: sc[b,h,l,r] += q[b,l,h,:] . dist_emb[l-r+1025] ----------
__global__ void k_qpe(const bf16* __restrict__ qb, const bf16* __restrict__ de,
                      float* __restrict__ sc, int tilesTotal) {
  int t = wave_id();
  if (t >= tilesTotal) return;
  const int rt = t & 63; t >>= 6;
  const int l = t & (NS - 1); const int b = t >> 10;
  const int lane = threadIdx.x & 31;
  const int r0 = rt << 4;
  v8f acc = {};
  for (int k0 = 0; k0 < NDH; k0 += 32) {
    int m = lane & 15; if (m >= NH) m = NH - 1;
    const int kba = k0 + ((lane & 16) ? 8 : 0);
    const bf16* pa = qb + (size_t)b * NS * ND + (size_t)l * ND + m * NDH + kba;
    v16bf a;
#pragma unroll
    for (int j = 0; j < 8; ++j) { a[j] = pa[j]; a[j + 8] = pa[j + 16]; }
    const int n = lane & 15;
    const int idx = l - (r0 + n) + (MAXP - 1);
    const int kbb = k0 + ((lane & 16) ? 8 : 0);
    const bf16* pb = de + (size_t)idx * NDH + kbb;
    v16bf bb;
#pragma unroll
    for (int j = 0; j < 8; ++j) { bb[j] = pb[j]; bb[j + 8] = pb[j + 16]; }
    acc = wmma_bf16(a, bb, acc);
  }
  const int n  = r0 + (lane & 15);
  const int mo = (lane & 16) ? 8 : 0;
#pragma unroll
  for (int g = 0; g < 8; ++g) {
    const int h = mo + g;
    if (h < NH) sc[(((size_t)(b * NH + h)) * NS + l) * NS + n] += acc[g];
  }
}

// ---------- finalize: sc = sc/8 + mask + attention_bias ----------
__global__ void k_scfin(float* __restrict__ sc, const float* __restrict__ mask,
                        const float* __restrict__ abias, size_t n) {
  const size_t i = (size_t)blockIdx.x * 256 + threadIdx.x;
  if (i >= n) return;
  const int ki = (int)(i & (NS - 1));
  const int b  = (int)(i / ((size_t)NH * NS * NS));
  sc[i] = sc[i] * 0.125f + mask[(size_t)b * NS + ki] + abias[i];
}

// ---------- row softmax -> bf16 probs ----------
__global__ void k_softmax(const float* __restrict__ sc, bf16* __restrict__ pb) {
  const int row = blockIdx.x, tid = threadIdx.x;
  const float* r = sc + (size_t)row * NS;
  __shared__ float red[256];
  float mx = -1e30f;
  for (int i = tid; i < NS; i += 256) mx = fmaxf(mx, r[i]);
  red[tid] = mx; __syncthreads();
  for (int s = 128; s; s >>= 1) { if (tid < s) red[tid] = fmaxf(red[tid], red[tid + s]); __syncthreads(); }
  mx = red[0]; __syncthreads();
  float sum = 0.f;
  for (int i = tid; i < NS; i += 256) sum += __expf(r[i] - mx);
  red[tid] = sum; __syncthreads();
  for (int s = 128; s; s >>= 1) { if (tid < s) red[tid] += red[tid + s]; __syncthreads(); }
  const float inv = 1.0f / red[0];
  for (int i = tid; i < NS; i += 256) pb[(size_t)row * NS + i] = (bf16)(__expf(r[i] - mx) * inv);
}

// ---------- ctx: probs @ V (V pre-transposed to [B,H,DH,S]) ----------
__global__ void k_ctx(const bf16* __restrict__ pb, const bf16* __restrict__ vT,
                      float* __restrict__ ctx, int tilesTotal) {
  int t = wave_id();
  if (t >= tilesTotal) return;
  const int nt = t & 3; t >>= 2;
  const int mt = t & 63; const int bh = t >> 6;
  const int b = bh / NH, h = bh % NH;
  const bf16* A = pb + (size_t)bh * NS * NS;
  const bf16* W = vT + (size_t)bh * NDH * NS;
  const int lane = threadIdx.x & 31;
  v8f acc = {};
  for (int k0 = 0; k0 < NS; k0 += 32) {
    v16bf a  = load_a_frag(A, NS, mt << 4, k0, lane);
    v16bf bb = load_b_frag(W, NS, nt << 4, k0, lane);
    acc = wmma_bf16(a, bb, acc);
  }
  const int n  = (nt << 4) + (lane & 15);
  const int mo = (mt << 4) + ((lane & 16) ? 8 : 0);
#pragma unroll
  for (int g = 0; g < 8; ++g)
    ctx[(size_t)b * NS * ND + (size_t)(mo + g) * ND + h * NDH + n] = acc[g];
}

// ---------- residual + LayerNorm (also emits bf16 copy) ----------
__global__ void k_ln_residual(const float* __restrict__ pj, const float* __restrict__ res,
                              const float* __restrict__ gg, const float* __restrict__ bb,
                              float* __restrict__ out, bf16* __restrict__ outb) {
  const int row = blockIdx.x, tid = threadIdx.x;
  const float* a = pj + (size_t)row * ND;
  const float* r = res + (size_t)row * ND;
  __shared__ float red[256];
  float s = 0.f;
  for (int i = tid; i < ND; i += 256) s += a[i] + r[i];
  red[tid] = s; __syncthreads();
  for (int st = 128; st; st >>= 1) { if (tid < st) red[tid] += red[tid + st]; __syncthreads(); }
  const float mu = red[0] / (float)ND; __syncthreads();
  float v = 0.f;
  for (int i = tid; i < ND; i += 256) { const float d = a[i] + r[i] - mu; v += d * d; }
  red[tid] = v; __syncthreads();
  for (int st = 128; st; st >>= 1) { if (tid < st) red[tid] += red[tid + st]; __syncthreads(); }
  const float rs = rsqrtf(red[0] / (float)ND + 1e-12f);
  for (int i = tid; i < ND; i += 256) {
    const float val = (a[i] + r[i] - mu) * rs * gg[i] + bb[i];
    out [(size_t)row * ND + i] = val;
    outb[(size_t)row * ND + i] = (bf16)val;
  }
}

// ---------- per-expert LN + gated accumulate into output ----------
__global__ void k_expert_out(const float* __restrict__ y, const float* __restrict__ ao,
                             const float* __restrict__ gates,
                             const float* __restrict__ gg, const float* __restrict__ bb,
                             float* __restrict__ out, int e, int accum) {
  const int row = blockIdx.x, tid = threadIdx.x;
  const float* a = y  + (size_t)row * ND;
  const float* r = ao + (size_t)row * ND;
  const float gate = gates[row * NE + e];
  __shared__ float red[256];
  float s = 0.f;
  for (int i = tid; i < ND; i += 256) s += a[i] + r[i];
  red[tid] = s; __syncthreads();
  for (int st = 128; st; st >>= 1) { if (tid < st) red[tid] += red[tid + st]; __syncthreads(); }
  const float mu = red[0] / (float)ND; __syncthreads();
  float v = 0.f;
  for (int i = tid; i < ND; i += 256) { const float d = a[i] + r[i] - mu; v += d * d; }
  red[tid] = v; __syncthreads();
  for (int st = 128; st; st >>= 1) { if (tid < st) red[tid] += red[tid + st]; __syncthreads(); }
  const float rs = rsqrtf(red[0] / (float)ND + 1e-12f);
  for (int i = tid; i < ND; i += 256) {
    const float val = gate * ((a[i] + r[i] - mu) * rs * gg[i] + bb[i]);
    if (accum) out[(size_t)row * ND + i] += val;
    else       out[(size_t)row * ND + i]  = val;
  }
}

// ---------- router: softmax(logits) -> top2 normalized gates ----------
__global__ void k_router(const float* __restrict__ ao, const float* __restrict__ rW,
                         const float* __restrict__ rb, float* __restrict__ gates) {
  const int t = blockIdx.x * blockDim.x + threadIdx.x;
  if (t >= NT) return;
  const float* a = ao + (size_t)t * ND;
  float lg[NE];
#pragma unroll
  for (int e = 0; e < NE; ++e) {
    float s = rb[e];
    const float* w = rW + (size_t)e * ND;
    for (int d = 0; d < ND; ++d) s += a[d] * w[d];
    lg[e] = s;
  }
  float mx = lg[0];
#pragma unroll
  for (int e = 1; e < NE; ++e) mx = fmaxf(mx, lg[e]);
  float sum = 0.f;
#pragma unroll
  for (int e = 0; e < NE; ++e) { lg[e] = __expf(lg[e] - mx); sum += lg[e]; }
  const float inv = 1.0f / sum;
  int i1 = 0;
#pragma unroll
  for (int e = 1; e < NE; ++e) if (lg[e] > lg[i1]) i1 = e;
  int i2 = (i1 == 0) ? 1 : 0;
#pragma unroll
  for (int e = 0; e < NE; ++e) if (e != i1 && lg[e] > lg[i2]) i2 = e;
  const float w1 = lg[i1] * inv, w2 = lg[i2] * inv, norm = 1.0f / (w1 + w2);
#pragma unroll
  for (int e = 0; e < NE; ++e)
    gates[t * NE + e] = (e == i1) ? w1 * norm : ((e == i2) ? w2 * norm : 0.0f);
}

// ---------- elementwise f32 -> bf16 ----------
__global__ void k_cvt(const float* __restrict__ s, bf16* __restrict__ d, size_t n) {
  const size_t i = (size_t)blockIdx.x * 256 + threadIdx.x;
  if (i < n) d[i] = (bf16)s[i];
}

// ---------- V transpose: [B,S,H,DH] -> [B,H,DH,S] ----------
__global__ void k_vT(const bf16* __restrict__ vb, bf16* __restrict__ vT, size_t n) {
  const size_t i = (size_t)blockIdx.x * 256 + threadIdx.x;
  if (i >= n) return;
  const int d = (int)(i % ND);
  const size_t rest = i / ND;
  const int s = (int)(rest % NS);
  const int b = (int)(rest / NS);
  const int h = d / NDH, dh = d % NDH;
  vT[(((size_t)(b * NH + h)) * NDH + dh) * NS + s] = vb[i];
}

extern "C" void kernel_launch(void* const* d_in, const int* in_sizes, int n_in,
                              void* d_out, int out_size, void* d_ws, size_t ws_size,
                              hipStream_t stream) {
  const float* x    = (const float*)d_in[0];
  const float* mask = (const float*)d_in[1];
  const float* ab   = (const float*)d_in[2];
  const float* Wq   = (const float*)d_in[3];
  const float* bq   = (const float*)d_in[4];
  const float* Wk   = (const float*)d_in[5];
  const float* bk   = (const float*)d_in[6];
  const float* Wv   = (const float*)d_in[7];
  const float* bv   = (const float*)d_in[8];
  const float* de   = (const float*)d_in[9];
  const float* Wo   = (const float*)d_in[10];
  const float* bo   = (const float*)d_in[11];
  const float* g1   = (const float*)d_in[12];
  const float* b1   = (const float*)d_in[13];
  const float* rW   = (const float*)d_in[14];
  const float* rb   = (const float*)d_in[15];
  const float* e1   = (const float*)d_in[16];
  const float* e2   = (const float*)d_in[17];
  const float* eo   = (const float*)d_in[18];
  const float* eg   = (const float*)d_in[19];
  const float* eb   = (const float*)d_in[20];

  float* out = (float*)d_out;
  float* sc  = out + (size_t)NB * NS * ND;   // attention_bias_out region

  char* p = (char*)d_ws;
  auto alloc = [&](size_t bytes) { char* r = p; p += (bytes + 255) & ~(size_t)255; return r; };
  bf16* xb    = (bf16*)alloc((size_t)NB * NS * ND * 2);
  bf16* wqb   = (bf16*)alloc((size_t)ND * ND * 2);
  bf16* wkb   = (bf16*)alloc((size_t)ND * ND * 2);
  bf16* wvb   = (bf16*)alloc((size_t)ND * ND * 2);
  bf16* wob   = (bf16*)alloc((size_t)ND * ND * 2);
  bf16* deb   = (bf16*)alloc((size_t)(2 * MAXP - 1) * NDH * 2);
  bf16* qb    = (bf16*)alloc((size_t)NT * ND * 2);
  bf16* kb    = (bf16*)alloc((size_t)NT * ND * 2);
  bf16* vb    = (bf16*)alloc((size_t)NT * ND * 2);
  bf16* vTb   = (bf16*)alloc((size_t)NB * NH * NDH * NS * 2);
  bf16* prb   = (bf16*)alloc((size_t)NB * NH * NS * NS * 2);
  float* aof  = (float*)alloc((size_t)NT * ND * 4);
  bf16* aob   = (bf16*)alloc((size_t)NT * ND * 2);
  bf16* e1b   = (bf16*)alloc((size_t)NE * NI * ND * 2);
  bf16* e2b   = (bf16*)alloc((size_t)NE * NI * ND * 2);
  bf16* eob   = (bf16*)alloc((size_t)NE * ND * NI * 2);
  bf16* gbuf  = (bf16*)alloc((size_t)NT * NI * 2);
  float* yf   = (float*)alloc((size_t)NT * ND * 4);
  float* gts  = (float*)alloc((size_t)NT * NE * 4);
  float* tmpF = (float*)alloc((size_t)NT * ND * 4);

  auto cvt = [&](const float* s, bf16* d, size_t n) {
    k_cvt<<<dim3((unsigned)((n + 255) / 256)), dim3(256), 0, stream>>>(s, d, n);
  };
  auto gemmT = [&](const bf16* A, int lda, const bf16* W, int ldw, float* C, int ldc,
                   const float* bias, int M, int N, int K) {
    k_gemm_tiled<<<dim3((M / BM) * (N / BN)), dim3(256), 0, stream>>>(
        A, lda, W, ldw, C, ldc, bias, N / BN, K);
  };

  // --- precision staging ---
  cvt(x, xb, (size_t)NB * NS * ND);
  cvt(Wq, wqb, (size_t)ND * ND); cvt(Wk, wkb, (size_t)ND * ND);
  cvt(Wv, wvb, (size_t)ND * ND); cvt(Wo, wob, (size_t)ND * ND);
  cvt(de, deb, (size_t)(2 * MAXP - 1) * NDH);
  cvt(e1, e1b, (size_t)NE * NI * ND);
  cvt(e2, e2b, (size_t)NE * NI * ND);
  cvt(eo, eob, (size_t)NE * ND * NI);

  // --- Q,K,V projections ---
  gemmT(xb, ND, wqb, ND, tmpF, ND, bq, NT, ND, ND); cvt(tmpF, qb, (size_t)NT * ND);
  gemmT(xb, ND, wkb, ND, tmpF, ND, bk, NT, ND, ND); cvt(tmpF, kb, (size_t)NT * ND);
  gemmT(xb, ND, wvb, ND, tmpF, ND, bv, NT, ND, ND); cvt(tmpF, vb, (size_t)NT * ND);
  k_vT<<<dim3((unsigned)(((size_t)NT * ND + 255) / 256)), dim3(256), 0, stream>>>(vb, vTb, (size_t)NT * ND);

  // --- attention scores (into d_out second region) ---
  {
    const int tiles = NB * NH * (NS / 16) * (NS / 16);
    k_scores<<<dim3((tiles + 3) / 4), dim3(128), 0, stream>>>(qb, kb, sc, tiles);
  }
  {
    const int tiles = NB * NS * (NS / 16);
    k_qpe<<<dim3((tiles + 3) / 4), dim3(128), 0, stream>>>(qb, deb, sc, tiles);
  }
  {
    const size_t n = (size_t)NB * NH * NS * NS;
    k_scfin<<<dim3((unsigned)((n + 255) / 256)), dim3(256), 0, stream>>>(sc, mask, ab, n);
  }
  k_softmax<<<dim3(NB * NH * NS), dim3(256), 0, stream>>>(sc, prb);

  // --- context + output projection + LN ---
  {
    const int tiles = NB * NH * (NS / 16) * (NDH / 16);
    k_ctx<<<dim3((tiles + 3) / 4), dim3(128), 0, stream>>>(prb, vTb, tmpF, tiles);
  }
  cvt(tmpF, qb, (size_t)NT * ND);                       // reuse qb as ctx_bf16
  gemmT(qb, ND, wob, ND, yf, ND, bo, NT, ND, ND);       // attn out-proj
  k_ln_residual<<<dim3(NT), dim3(256), 0, stream>>>(yf, x, g1, b1, aof, aob);

  // --- router gates ---
  k_router<<<dim3((NT + 255) / 256), dim3(256), 0, stream>>>(aof, rW, rb, gts);

  // --- dense MoE (all experts, matching reference) ---
  for (int e = 0; e < NE; ++e) {
    k_moe_h_tiled<<<dim3((NT / BM) * (NI / BN)), dim3(256), 0, stream>>>(
        aob, e1b + (size_t)e * NI * ND, e2b + (size_t)e * NI * ND, gbuf);
    gemmT(gbuf, NI, eob + (size_t)e * ND * NI, NI, yf, ND, nullptr, NT, ND, NI);
    k_expert_out<<<dim3(NT), dim3(256), 0, stream>>>(
        yf, aof, gts, eg + (size_t)e * ND, eb + (size_t)e * ND, out, e, e > 0);
  }
}
